// LinearAttention_4002909520010
// MI455X (gfx1250) — matrix operand order repaired and checked by reading, not compiled, not run on hardware
//
#include <hip/hip_runtime.h>

// ---------------------------------------------------------------------------
// LinearAttention (B=32, N=4096, C=256, 8 heads, d=32) for MI455X / gfx1250.
// Memory-bound (~1.2 GB traffic vs 40 GFLOP) -> bf16 WMMA, bf16 intermediates,
// hardware packed f32->bf16 converts, TDM async tensor load into LDS.
// Workspace requirement: ~207 MB.
// ---------------------------------------------------------------------------

typedef unsigned short u16;
typedef unsigned int   u32;

typedef __attribute__((ext_vector_type(16))) __bf16 v16bf;
typedef __attribute__((ext_vector_type(2)))  __bf16 bf16x2;
typedef __attribute__((ext_vector_type(8)))  float  v8f;
typedef __attribute__((ext_vector_type(2)))  float  f32x2;
typedef __attribute__((ext_vector_type(4)))  unsigned int u32x4;
typedef __attribute__((ext_vector_type(8)))  int    i32x8;
typedef __attribute__((ext_vector_type(4)))  int    i32x4;

union BF16Frag { v16bf v; u16 u[16]; u32 d[8]; };
union F32Frag  { v8f v; float f[8]; };

#define BATCH 32
#define NTOK  4096
#define CDIM  256
#define NH    8
#define HD    32
#define ROWS  (BATCH * NTOK)   // 131072

#if __has_builtin(__builtin_amdgcn_tensor_load_to_lds)
#define USE_TDM 1
#else
#define USE_TDM 0
#endif

// Hardware converts: f32 -> bf16 (single + packed pair -> one dword).
__device__ __forceinline__ u16 f32_to_bf16(float f) {
  union { __bf16 h; u16 u; } c; c.h = (__bf16)f; return c.u;
}
__device__ __forceinline__ u32 pack2_bf16(float lo, float hi) {
  union { bf16x2 h; u32 w; } c;
  c.h = __builtin_convertvector((f32x2){lo, hi}, bf16x2);
  return c.w;
}
__device__ __forceinline__ float bf16_to_f32(u16 h) {
  return __uint_as_float(((u32)h) << 16);
}
// Cross-lane XOR-1 exchange: ds_swizzle group-of-32, xor=1, and=0x1f (SWAPX1).
// EXEC must be all-ones at the call site (we only call under wave-uniform
// control flow).
__device__ __forceinline__ float lane_xor1(float v) {
  return __int_as_float(__builtin_amdgcn_ds_swizzle(__float_as_int(v), 0x041F));
}
// 16-bit A fragment K index for VGPR-pair j (0..7), lane half (0/1):
// lanes0-15: K=0,1 / 2,3 / 4,5 / 6,7 / 16..23 ; lanes16-31: 8..15 / 24..31
__device__ __forceinline__ int a_kidx(int j, int half) {
  return ((j >> 2) << 4) + (half << 3) + ((j & 3) << 1);
}

// ---------------------------------------------------------------------------
// Kernel 0: one-time f32 -> bf16 conversion of Wqk (512x256, L2 resident).
// ---------------------------------------------------------------------------
__global__ __launch_bounds__(256) void k_cvt_w(
    const float* __restrict__ Wqk, u16* __restrict__ w_bf)
{
  const int i = blockIdx.x * 256 + threadIdx.x;    // [0, 65536): 2 elems each
  const float2 f2 = *(const float2*)(Wqk + 2 * i);
  *(u32*)(w_bf + 2 * i) = pack2_bf16(f2.x, f2.y);
}

// ---------------------------------------------------------------------------
// Kernel 1: qk = x @ Wqk^T + b ; q/k = elu+1 ; RoPE ; k-sum for normalizer.
// Each wave computes a 16(M) x 64(N) output tile, K=256 in chunks of 32.
// blockIdx.y < 4 -> q-channel blocks; >= 4 -> k-channel blocks (uniform!).
// ---------------------------------------------------------------------------
__global__ __launch_bounds__(256) void k_proj_rope(
    const float* __restrict__ x, const u16* __restrict__ w_bf,
    const float* __restrict__ bqk,
    u16* __restrict__ q_rope, u16* __restrict__ k_rope,
    u16* __restrict__ q_plain, float* __restrict__ ksum)
{
  const int lane = threadIdx.x & 31;
  const int wave = threadIdx.x >> 5;
  const int half = lane >> 4;
  const int lcol = lane & 15;
  const int r0 = blockIdx.x * 128 + wave * 16;   // 16 rows per wave
  const int c0 = blockIdx.y * 64;                // 64 output channels per wave

  F32Frag acc[4];
  #pragma unroll
  for (int t = 0; t < 4; ++t)
    #pragma unroll
    for (int i = 0; i < 8; ++i) acc[t].f[i] = 0.0f;

  const float* xrow = x + (size_t)(r0 + lcol) * CDIM;

  for (int kk = 0; kk < CDIM; kk += 32) {
    BF16Frag a;
    #pragma unroll
    for (int j = 0; j < 8; ++j) {
      const int K = kk + half * 16 + 2 * j;
      const float2 f2 = *(const float2*)(xrow + K);
      a.d[j] = pack2_bf16(f2.x, f2.y);           // v_cvt_pk_bf16_f32
    }
    #pragma unroll
    for (int t = 0; t < 4; ++t) {
      // B[k, n] = Wqk[c0+t*16+n, k] : pre-converted bf16, contiguous per lane.
      const u16* wrow = w_bf + (size_t)(c0 + t * 16 + lcol) * CDIM;
      BF16Frag b;
      #pragma unroll
      for (int j = 0; j < 8; ++j)
        b.d[j] = *(const u32*)(wrow + kk + half * 16 + 2 * j);
      acc[t].v = __builtin_amdgcn_wmma_f32_16x16x32_bf16(
          false, a.v, false, b.v, (short)0, acc[t].v, false, false);
    }
  }

  const int batch = r0 >> 12;            // r0 / 4096 (tile never crosses batch)
  const bool is_q = (blockIdx.y < 4);    // WAVE-UNIFORM: scalar branch below
  #pragma unroll
  for (int t = 0; t < 4; ++t) {
    const int col = c0 + t * 16 + lcol;
    const int ch  = is_q ? col : (col - CDIM);
    const float bias = bqk[col];
    const int p  = ch >> 1;                                   // rope pair index
    const float theta = __expf(-(float)(p & 63) * 0.14391156831f); // ln(1e4)/64
    const bool use_h = (p < 64);
    const int odd = lcol & 1;            // (re,im) pair in adjacent lanes
    if (is_q) {
      #pragma unroll
      for (int r = 0; r < 8; ++r) {
        const int row  = r0 + r + 8 * half;     // C/D layout: M = r + 8*half
        const int nidx = row & (NTOK - 1);
        float val = acc[t].f[r] + bias;
        val = (val > 0.0f) ? (val + 1.0f) : __expf(val);   // elu(x)+1
        q_plain[(size_t)row * CDIM + ch] = f32_to_bf16(val);
        const float pos = (float)(use_h ? (nidx >> 6) : (nidx & 63));
        float s, c;
        __sincosf(pos * theta, &s, &c);
        const float other = lane_xor1(val);
        const float rot = odd ? (c * val + s * other)   // im' = c*im + s*re
                              : (c * val - s * other);  // re' = c*re - s*im
        q_rope[(size_t)row * CDIM + ch] = f32_to_bf16(rot);
      }
    } else {
      float ksum_part = 0.0f;
      #pragma unroll
      for (int r = 0; r < 8; ++r) {
        const int row  = r0 + r + 8 * half;
        const int nidx = row & (NTOK - 1);
        float val = acc[t].f[r] + bias;
        val = (val > 0.0f) ? (val + 1.0f) : __expf(val);   // elu(x)+1
        ksum_part += val;
        const float pos = (float)(use_h ? (nidx >> 6) : (nidx & 63));
        float s, c;
        __sincosf(pos * theta, &s, &c);
        const float other = lane_xor1(val);
        const float rot = odd ? (c * val + s * other)
                              : (c * val - s * other);
        k_rope[(size_t)row * CDIM + ch] = f32_to_bf16(rot);
      }
      atomicAdd(&ksum[batch * CDIM + ch], ksum_part);
    }
  }
}

// ---------------------------------------------------------------------------
// Kernel 2: kv[b,h] = (k_rope*s)^T @ (v*s), 32x32, K = N = 4096.
// k tile arrives via the Tensor Data Mover (async DMA into LDS, TENSORcnt);
// v tile is loaded + converted manually (f32 source).  4 waves/block, one
// 16x16 tile each; 128 WMMAs per wave.
// ---------------------------------------------------------------------------
__global__ __launch_bounds__(128) void k_kv(
    const float* __restrict__ x, const u16* __restrict__ k_rope,
    float* __restrict__ kv)
{
  __shared__ u16 kRaw[64][HD];   // [n_local][d]  (row-major, TDM layout)
  __shared__ u16 vT[HD][64];     // [e][n_local]  (transposed for B frags)
  const int bh = blockIdx.x;
  const int b = bh >> 3, h = bh & 7;
  const int tid  = threadIdx.x;
  const int lane = tid & 31, wave = tid >> 5;
  const int half = lane >> 4, lcol = lane & 15;
  const int d0 = (wave >> 1) * 16, e0 = (wave & 1) * 16;

  F32Frag acc;
  #pragma unroll
  for (int i = 0; i < 8; ++i) acc.f[i] = 0.0f;

  const int nl = tid >> 1;           // 0..63 : n within chunk
  const int cb = (tid & 1) * 16;     // channel half
  const size_t rowbase = (size_t)b * NTOK;

  for (int n0 = 0; n0 < NTOK; n0 += 64) {
#if USE_TDM
    if (wave == 0) {
      // Tensor DMA: 64 x 32 bf16 tile of k_rope -> LDS kRaw (row-major).
      const unsigned long long ga =
          (unsigned long long)(uintptr_t)(k_rope + (rowbase + n0) * CDIM + h * HD);
      const u32 lds_off = (u32)(uintptr_t)(&kRaw[0][0]);   // LDS addr = low 32b
      u32x4 g0;
      g0[0] = 1u;                                  // count=1, user descriptor
      g0[1] = lds_off;                             // lds_addr
      g0[2] = (u32)(ga & 0xffffffffu);             // global_addr[31:0]
      g0[3] = (u32)((ga >> 32) & 0x01ffffffu) | 0x80000000u; // addr[56:32]|type=2
      i32x8 g1;
      g1[0] = (1 << 16);          // data_size=2 bytes, no flags
      g1[1] = (HD << 16);         // tensor_dim0 = 32 (lo16 in [31:16])
      g1[2] = (64 << 16);         // tensor_dim1 = 64 (lo16 in [31:16])
      g1[3] = (HD << 16);         // tile_dim0 = 32
      g1[4] = 64;                 // tile_dim1 = 64, tile_dim2 = 0
      g1[5] = CDIM;               // tensor_dim0_stride = 256 elements
      g1[6] = 0;
      g1[7] = 0;
      const i32x4 gz4 = {0, 0, 0, 0};
      const i32x8 gz8 = {0, 0, 0, 0, 0, 0, 0, 0};
      __builtin_amdgcn_tensor_load_to_lds(g0, g1, gz4, gz4, gz8, 0);
    }
#else
    {
      const u32* ks = (const u32*)(k_rope + (rowbase + n0 + nl) * CDIM + h * HD + cb);
      #pragma unroll
      for (int i = 0; i < 8; ++i)
        *(u32*)&kRaw[nl][cb + 2 * i] = ks[i];
    }
#endif
    // v tile: load f32, convert each to bf16, store transposed.
    const float* vs = x + (rowbase + n0 + nl) * CDIM + h * HD + cb;
    #pragma unroll
    for (int i = 0; i < 16; ++i) vT[cb + i][nl] = f32_to_bf16(vs[i]);
#if USE_TDM
    if (wave == 0) __builtin_amdgcn_s_wait_tensorcnt(0);
#endif
    __syncthreads();

    #pragma unroll
    for (int sub = 0; sub < 2; ++sub) {
      BF16Frag a, bfr;
      #pragma unroll
      for (int j = 0; j < 8; ++j) {              // A[m,k] = kRaw[k][d0+m]
        const int K = sub * 32 + half * 16 + 2 * j;
        a.u[2 * j]     = kRaw[K][d0 + lcol];
        a.u[2 * j + 1] = kRaw[K + 1][d0 + lcol];
      }
      #pragma unroll
      for (int j = 0; j < 8; ++j)                // B[k,e] = vT[e][k] contiguous
        bfr.d[j] = *(const u32*)&vT[e0 + lcol][sub * 32 + half * 16 + 2 * j];
      acc.v = __builtin_amdgcn_wmma_f32_16x16x32_bf16(
          false, a.v, false, bfr.v, (short)0, acc.v, false, false);
    }
    __syncthreads();
  }
  #pragma unroll
  for (int r = 0; r < 8; ++r) {
    const int M = r + 8 * half;
    kv[(size_t)bh * (HD * HD) + (d0 + M) * HD + e0 + lcol] =
        acc.f[r] * (1.0f / 4096.0f);   // scale^2 = 1/N
  }
}

// ---------------------------------------------------------------------------
// Kernel 3: z[b,h,n] = 1 / (q_plain . (ksum/N) + 1e-6)
// ---------------------------------------------------------------------------
__global__ __launch_bounds__(256) void k_z(
    const u16* __restrict__ q_plain, const float* __restrict__ ksum,
    float* __restrict__ z)
{
  const int tid = blockIdx.x * 256 + threadIdx.x;   // [0, B*NH*NTOK)
  const int n = tid & (NTOK - 1);
  const int h = (tid >> 12) & 7;
  const int b = tid >> 15;
  const u32* q2 = (const u32*)(q_plain + ((size_t)(b * NTOK + n)) * CDIM + h * HD);
  const float* ks = ksum + b * CDIM + h * HD;
  float dot = 0.0f;
  #pragma unroll
  for (int i = 0; i < 16; ++i) {
    const u32 w2 = q2[i];
    dot += bf16_to_f32((u16)(w2 & 0xffffu)) * ks[2 * i]
         + bf16_to_f32((u16)(w2 >> 16))     * ks[2 * i + 1];
  }
  z[tid] = 1.0f / (dot * (1.0f / 4096.0f) + 1e-6f);
}

// ---------------------------------------------------------------------------
// Kernel 4: out = (q_rope @ kv) * z.  K=32 -> one WMMA per 16x16 tile.
// ---------------------------------------------------------------------------
__global__ __launch_bounds__(256) void k_out(
    const u16* __restrict__ q_rope, const float* __restrict__ kv,
    const float* __restrict__ z, float* __restrict__ out)
{
  const int lane = threadIdx.x & 31;
  const int wave = threadIdx.x >> 5;
  const int half = lane >> 4, lcol = lane & 15;
  const int r0 = blockIdx.x * 128 + wave * 16;
  const int h  = blockIdx.y;
  const int b  = r0 >> 12;

  BF16Frag a;
  const u16* qr = q_rope + (size_t)(r0 + lcol) * CDIM + h * HD;
  #pragma unroll
  for (int j = 0; j < 8; ++j) a.d[j] = *(const u32*)(qr + a_kidx(j, half));

  const float* kvp = kv + (size_t)(b * NH + h) * (HD * HD);
  const float* zp  = z  + (size_t)(b * NH + h) * NTOK;

  #pragma unroll
  for (int tj = 0; tj < 2; ++tj) {
    const int e0 = tj * 16;
    BF16Frag bfr;
    #pragma unroll
    for (int j = 0; j < 8; ++j) {
      const int K = a_kidx(j, half);
      bfr.d[j] = pack2_bf16(kvp[K * HD + e0 + lcol], kvp[(K + 1) * HD + e0 + lcol]);
    }
    F32Frag acc;
    #pragma unroll
    for (int i = 0; i < 8; ++i) acc.f[i] = 0.0f;
    acc.v = __builtin_amdgcn_wmma_f32_16x16x32_bf16(
        false, a.v, false, bfr.v, (short)0, acc.v, false, false);
    #pragma unroll
    for (int r = 0; r < 8; ++r) {
      const int M = r + 8 * half;
      const int row = r0 + M;
      out[(size_t)row * CDIM + h * HD + e0 + lcol] = acc.f[r] * zp[row & (NTOK - 1)];
    }
  }
}

// ---------------------------------------------------------------------------
// Kernel 5: LePE depthwise 3x3 conv on v (= x), accumulated into out.
// ---------------------------------------------------------------------------
__global__ __launch_bounds__(256) void k_lepe(
    const float* __restrict__ x, const float* __restrict__ w,
    const float* __restrict__ bias, float* __restrict__ out)
{
  const int idx = blockIdx.x * 256 + threadIdx.x;   // [0, B*N*C)
  const int c = idx & (CDIM - 1);
  const int n = (idx >> 8) & (NTOK - 1);
  const int b = idx >> 20;
  const int ih = n >> 6, iw = n & 63;
  float s = bias[c];
  const float* wp = w + c * 9;
  const size_t base = (size_t)b * NTOK * CDIM;
  // Warm L2 for the row below (global_prefetch_b8).
  if (ih < 63) __builtin_prefetch(&x[base + (size_t)(n + 64) * CDIM + c], 0, 1);
  #pragma unroll
  for (int dy = -1; dy <= 1; ++dy) {
    const int y = ih + dy;
    if ((unsigned)y >= 64u) continue;
    #pragma unroll
    for (int dx = -1; dx <= 1; ++dx) {
      const int xx = iw + dx;
      if ((unsigned)xx >= 64u) continue;
      s += wp[(dy + 1) * 3 + (dx + 1)] * x[base + (size_t)(y * 64 + xx) * CDIM + c];
    }
  }
  out[idx] += s;
}

// ---------------------------------------------------------------------------
extern "C" void kernel_launch(void* const* d_in, const int* in_sizes, int n_in,
                              void* d_out, int out_size, void* d_ws, size_t ws_size,
                              hipStream_t stream) {
  (void)in_sizes; (void)n_in; (void)out_size; (void)ws_size;
  const float* x      = (const float*)d_in[0];
  const float* Wqk    = (const float*)d_in[1];
  const float* bqk    = (const float*)d_in[2];
  const float* lepe_w = (const float*)d_in[3];
  const float* lepe_b = (const float*)d_in[4];
  float* out = (float*)d_out;

  char* ws = (char*)d_ws;
  const size_t ropeBytes = (size_t)ROWS * CDIM * sizeof(u16);      // 64 MB each
  u16*   q_rope  = (u16*)(ws);
  u16*   k_rope  = (u16*)(ws + ropeBytes);
  u16*   q_plain = (u16*)(ws + 2 * ropeBytes);
  size_t off = 3 * ropeBytes;
  float* ksum = (float*)(ws + off); off += (size_t)BATCH * CDIM * 4;
  float* kv   = (float*)(ws + off); off += (size_t)BATCH * NH * HD * HD * 4;
  float* z    = (float*)(ws + off); off += (size_t)BATCH * NH * NTOK * 4;
  u16*   w_bf = (u16*)(ws + off);   off += (size_t)2 * CDIM * CDIM * 2;

  (void)hipMemsetAsync(ksum, 0, (size_t)BATCH * CDIM * sizeof(float), stream);
  k_cvt_w<<<(2 * CDIM * CDIM / 2) / 256, 256, 0, stream>>>(Wqk, w_bf);
  k_proj_rope<<<dim3(ROWS / 128, 8), 256, 0, stream>>>(x, w_bf, bqk, q_rope,
                                                       k_rope, q_plain, ksum);
  k_kv<<<BATCH * NH, 128, 0, stream>>>(x, k_rope, kv);
  k_z<<<(BATCH * NH * NTOK) / 256, 256, 0, stream>>>(q_plain, ksum, z);
  k_out<<<dim3(ROWS / 128, NH), 256, 0, stream>>>(q_rope, kv, z, out);
  k_lepe<<<(ROWS * CDIM) / 256, 256, 0, stream>>>(x, lepe_w, lepe_b, out);
}
